// MSGFusionNet_54726473286076
// MI455X (gfx1250) — matrix operand order repaired and checked by reading, not compiled, not run on hardware
//
#include <hip/hip_runtime.h>
#include <hip/hip_bf16.h>
#include <cstdint>

typedef __attribute__((ext_vector_type(16))) _Float16 v16h;
typedef __attribute__((ext_vector_type(8)))  _Float16 v8h;
typedef __attribute__((ext_vector_type(8)))  float    v8f;
typedef int v4i __attribute__((vector_size(16)));   // matches builtin's V4i

#define DEV __device__ __forceinline__

// ---------------------------------------------------------------------------
// CDNA5 async global->LDS staging (ASYNCcnt-tracked).
// ---------------------------------------------------------------------------
#if defined(__has_builtin)
# if __has_builtin(__builtin_amdgcn_global_load_async_to_lds_b128)
#  define HAVE_ASYNC_LDS 1
# endif
#endif
#ifndef HAVE_ASYNC_LDS
# define HAVE_ASYNC_LDS 0
#endif

#if HAVE_ASYNC_LDS
DEV void async_copy_b128(void* lds, const void* g) {
  __builtin_amdgcn_global_load_async_to_lds_b128(
      (__attribute__((address_space(1))) v4i*)(g),
      (__attribute__((address_space(3))) v4i*)(lds), 0, 0);
}
DEV void wait_async() {
# if __has_builtin(__builtin_amdgcn_s_wait_asynccnt)
  __builtin_amdgcn_s_wait_asynccnt(0);
# else
  asm volatile("s_wait_asynccnt 0x0" ::: "memory");
# endif
}
#endif

// ---------------------------------------------------------------------------
// WMMA fragment loaders (CDNA5 ISA 7.12.2 layouts, wave32)
// A: 16x32 f16 from row-major [M][K]; B: 32x16 f16 from row-major [N][K];
// C/D: 16x16 f32, lane n = L&15, VGPR r -> m = (L>>4)*8 + r.
// All offsets are 16B aligned -> two ds_load_b128 per fragment.
// ---------------------------------------------------------------------------
DEV v16h ldA_frag(const _Float16* base, int ldk, int m0, int k0) {
  int lane = threadIdx.x & 31;
  int m = m0 + (lane & 15);
  int hi = lane >> 4;
  const _Float16* p = base + m * ldk + k0 + hi * 8;
  union { v16h v; v8h h[2]; } u;
  u.h[0] = *(const v8h*)p;
  u.h[1] = *(const v8h*)(p + 16);
  return u.v;
}

DEV v16h ldB_frag(const _Float16* base, int ldk, int n0, int k0) {
  int lane = threadIdx.x & 31;
  int n = n0 + (lane & 15);
  int hi = lane >> 4;
  const _Float16* p = base + n * ldk + k0 + hi * 8;
  union { v16h v; v8h h[2]; } u;
  u.h[0] = *(const v8h*)p;
  u.h[1] = *(const v8h*)(p + 16);
  return u.v;
}

DEV v8f wmma_f16(v16h a, v16h b, v8f c) {
  return __builtin_amdgcn_wmma_f32_16x16x32_f16(false, a, false, b, (short)0, c,
                                                false, false);
}

// ---------------------------------------------------------------------------
// Constants
// ---------------------------------------------------------------------------
constexpr int C3   = 288;      // 3*DIM
constexpr int HW   = 65536;    // H*W
constexpr int NB   = 4;        // batch
constexpr int NWIN = 4096;     // B * 32 * 32 windows
constexpr float SCALE = 0.17677669529663687f; // 32^-0.5

// ---------------------------------------------------------------------------
// K1: relative-position bias MLP.  bias[h][i][j], i,j in [0,64)
// ---------------------------------------------------------------------------
__global__ void bias_mlp(const float* __restrict__ mw1, const float* __restrict__ mb1,
                         const float* __restrict__ mw2, const float* __restrict__ mb2,
                         float* __restrict__ biasOut) {
  int idx = blockIdx.x * 256 + threadIdx.x;
  if (idx >= 4096) return;
  int i = idx >> 6, j = idx & 63;
  float d0 = (float)((i >> 3) - (j >> 3));
  float d1 = (float)((i & 7) - (j & 7));
  float s0 = (d0 < 0.f) ? -1.f : (d0 > 0.f ? 1.f : 0.f);
  float s1 = (d1 < 0.f) ? -1.f : (d1 > 0.f ? 1.f : 0.f);
  float r0 = s0 * log1pf(fabsf(d0));
  float r1 = s1 * log1pf(fabsf(d1));
  float o0 = 0.f, o1 = 0.f, o2 = 0.f;
  for (int k = 0; k < 256; ++k) {
    float hsum = fmaxf(r0 * mw1[k] + r1 * mw1[256 + k] + mb1[k], 0.f);
    o0 += hsum * mw2[k * 3 + 0];
    o1 += hsum * mw2[k * 3 + 1];
    o2 += hsum * mw2[k * 3 + 2];
  }
  biasOut[0 * 4096 + idx] = o0 + mb2[0];
  biasOut[1 * 4096 + idx] = o1 + mb2[1];
  biasOut[2 * 4096 + idx] = o2 + mb2[2];
}

// ---------------------------------------------------------------------------
// K2: QKV conv1x1 (main branch).  One block = one 8x8 window.
// GEMM [64x96] x [96->288], OC in 2 halves of 144 (LDS < 64KB).
// A-fragments hoisted per wave; B-fragments software-pipelined; the V-plane
// side-store condition is wave-uniform (half==1 && nt>=3  <=>  oc>=192).
// ---------------------------------------------------------------------------
__global__ __launch_bounds__(256) void conv_qkv_main(
    const float* __restrict__ x, const float* __restrict__ wqk,
    const float* __restrict__ bqk, const float* __restrict__ wv,
    const float* __restrict__ bv, _Float16* __restrict__ qkvh,
    _Float16* __restrict__ vh) {
  __shared__ __align__(16) _Float16 sx[64 * 96];
  __shared__ __align__(16) _Float16 sw[144 * 96];
  __shared__ float sb[144];
  int tid = threadIdx.x;
  int win = blockIdx.x;
  int b = win >> 10, wy = (win >> 5) & 31, wx = win & 31;

  for (int e = tid; e < 64 * 96; e += 256) {
    int tok = e / 96, c = e - tok * 96;
    int hh = wy * 8 + (tok >> 3), ww = wx * 8 + (tok & 7);
    sx[e] = (_Float16)x[(((size_t)b * 96 + c) << 16) + (hh << 8) + ww];
  }
  int wid = tid >> 5, lane = tid & 31, n = lane & 15, hi = lane >> 4;
  int mt = wid & 3;
  int ntLo = (wid < 4) ? 0 : 5;
  int ntHi = (wid < 4) ? 5 : 9;

  for (int half = 0; half < 2; ++half) {
    __syncthreads();
    for (int e = tid; e < 144 * 96; e += 256) {
      int o = e / 96, c = e - o * 96;
      int oc = half * 144 + o;
      sw[e] = (_Float16)((oc < 192) ? wqk[oc * 96 + c] : wv[(oc - 192) * 96 + c]);
    }
    for (int e = tid; e < 144; e += 256) {
      int oc = half * 144 + e;
      sb[e] = (oc < 192) ? bqk[oc] : bv[oc - 192];
    }
    __syncthreads();
    v16h a0 = ldA_frag(sx, 96, mt * 16, 0);
    v16h a1 = ldA_frag(sx, 96, mt * 16, 32);
    v16h a2 = ldA_frag(sx, 96, mt * 16, 64);
    v16h b0 = ldB_frag(sw, 96, ntLo * 16, 0);
    v16h b1 = ldB_frag(sw, 96, ntLo * 16, 32);
    v16h b2 = ldB_frag(sw, 96, ntLo * 16, 64);
    for (int nt = ntLo; nt < ntHi; ++nt) {
      v16h c0 = b0, c1 = b1, c2 = b2;
      if (nt + 1 < ntHi) {                       // prefetch next tile's B
        b0 = ldB_frag(sw, 96, (nt + 1) * 16, 0);
        b1 = ldB_frag(sw, 96, (nt + 1) * 16, 32);
        b2 = ldB_frag(sw, 96, (nt + 1) * 16, 64);
      }
      v8f acc = {};
      acc = wmma_f16(a0, c0, acc);
      acc = wmma_f16(a1, c1, acc);
      acc = wmma_f16(a2, c2, acc);
      int oc = half * 144 + nt * 16 + n;
      float bias = sb[nt * 16 + n];
      for (int r = 0; r < 8; ++r) {
        int tok = mt * 16 + hi * 8 + r;
        qkvh[((size_t)win * 64 + tok) * C3 + oc] = (_Float16)(acc[r] + bias);
      }
      if (half == 1 && nt >= 3) {                // uniform: whole tile is V
        int vc = oc - 192;
        for (int r = 0; r < 8; ++r) {
          int tok = mt * 16 + hi * 8 + r;
          int hh = wy * 8 + (tok >> 3), ww = wx * 8 + (tok & 7);
          vh[(((size_t)b * 96 + vc) << 16) + (hh << 8) + ww] =
              (_Float16)(acc[r] + bias);
        }
      }
    }
  }
}

// ---------------------------------------------------------------------------
// K3: aV = conv1x1(ass_vision, wav) + bav  (ass attention is dead code in the
// reference; only aV feeds the output).
// ---------------------------------------------------------------------------
__global__ __launch_bounds__(256) void conv_v_ass(
    const float* __restrict__ x, const float* __restrict__ wav,
    const float* __restrict__ bav, _Float16* __restrict__ avh) {
  __shared__ __align__(16) _Float16 sx[64 * 96];
  __shared__ __align__(16) _Float16 sw[96 * 96];
  __shared__ float sb[96];
  int tid = threadIdx.x;
  int win = blockIdx.x;
  int b = win >> 10, wy = (win >> 5) & 31, wx = win & 31;

  for (int e = tid; e < 64 * 96; e += 256) {
    int tok = e / 96, c = e - tok * 96;
    int hh = wy * 8 + (tok >> 3), ww = wx * 8 + (tok & 7);
    sx[e] = (_Float16)x[(((size_t)b * 96 + c) << 16) + (hh << 8) + ww];
  }
  for (int e = tid; e < 96 * 96; e += 256) sw[e] = (_Float16)wav[e];
  if (tid < 96) sb[tid] = bav[tid];
  __syncthreads();

  int wid = tid >> 5, lane = tid & 31, n = lane & 15, hi = lane >> 4;
  int mt = wid & 3;
  int ntLo = (wid >> 2) * 3;                 // waves 0-3: nt 0..2, 4-7: nt 3..5
  v16h a0 = ldA_frag(sx, 96, mt * 16, 0);
  v16h a1 = ldA_frag(sx, 96, mt * 16, 32);
  v16h a2 = ldA_frag(sx, 96, mt * 16, 64);
  v16h b0 = ldB_frag(sw, 96, ntLo * 16, 0);
  v16h b1 = ldB_frag(sw, 96, ntLo * 16, 32);
  v16h b2 = ldB_frag(sw, 96, ntLo * 16, 64);
#pragma unroll
  for (int k = 0; k < 3; ++k) {
    int nt = ntLo + k;
    v16h c0 = b0, c1 = b1, c2 = b2;
    if (k < 2) {
      b0 = ldB_frag(sw, 96, (nt + 1) * 16, 0);
      b1 = ldB_frag(sw, 96, (nt + 1) * 16, 32);
      b2 = ldB_frag(sw, 96, (nt + 1) * 16, 64);
    }
    v8f acc = {};
    acc = wmma_f16(a0, c0, acc);
    acc = wmma_f16(a1, c1, acc);
    acc = wmma_f16(a2, c2, acc);
    int oc = nt * 16 + n;
    float bias = sb[oc];
    for (int r = 0; r < 8; ++r) {
      int tok = mt * 16 + hi * 8 + r;
      int hh = wy * 8 + (tok >> 3), ww = wx * 8 + (tok & 7);
      avh[(((size_t)b * 96 + oc) << 16) + (hh << 8) + ww] = (_Float16)(acc[r] + bias);
    }
  }
}

// ---------------------------------------------------------------------------
// K4: window attention.  One block = one window; loop over 3 heads.
// HD=32 => each 16x16 S-tile is exactly one 16x16x32 WMMA.
// ---------------------------------------------------------------------------
__global__ __launch_bounds__(256) void win_attn(
    const _Float16* __restrict__ qkvh, const float* __restrict__ biasG,
    _Float16* __restrict__ attnh) {
  __shared__ __align__(16) _Float16 Qs[64 * 32];  // [tok][c]
  __shared__ __align__(16) _Float16 Ks[64 * 32];  // [tok][c] == B row-major [N][K]
  __shared__ __align__(16) _Float16 Vt[32 * 64];  // [c][tok] == B row-major [N][K]
  __shared__ float    S[64 * 64];
  __shared__ __align__(16) _Float16 P[64 * 64];

  int tid = threadIdx.x;
  int win = blockIdx.x;
  int b = win >> 10, wy = (win >> 5) & 31, wx = win & 31;
  int wid = tid >> 5, lane = tid & 31, n = lane & 15, hi = lane >> 4;
  const _Float16* g = qkvh + (size_t)win * 64 * C3;

  for (int h = 0; h < 3; ++h) {
    __syncthreads();                    // protect LDS reuse across heads
    {
      int tok = tid >> 2, q4 = tid & 3;          // 256 threads -> 64 x 4x16B
#if HAVE_ASYNC_LDS
      async_copy_b128(&Qs[tok * 32 + q4 * 8], g + tok * C3 + h * 32 + q4 * 8);
      async_copy_b128(&Ks[tok * 32 + q4 * 8], g + tok * C3 + 96 + h * 32 + q4 * 8);
#else
      const uint4* qsrc = (const uint4*)(g + tok * C3 + h * 32);
      const uint4* ksrc = (const uint4*)(g + tok * C3 + 96 + h * 32);
      ((uint4*)Qs)[tok * 4 + q4] = qsrc[q4];
      ((uint4*)Ks)[tok * 4 + q4] = ksrc[q4];
#endif
    }
    for (int e = tid; e < 32 * 64; e += 256) {   // V transposed: [c][tok]
      int c = e >> 6, tok = e & 63;
      Vt[e] = g[tok * C3 + 192 + h * 32 + c];
    }
#if HAVE_ASYNC_LDS
    wait_async();
#endif
    __syncthreads();

    // S tiles: wave owns mt = wid&3 and two nt; A reused, B pipelined.
    {
      int mt = wid & 3, ntBase = (wid >> 2) * 2;
      v16h a  = ldA_frag(Qs, 32, mt * 16, 0);
      v16h bA = ldB_frag(Ks, 32, ntBase * 16, 0);
      v16h bB = ldB_frag(Ks, 32, (ntBase + 1) * 16, 0);
#pragma unroll
      for (int k = 0; k < 2; ++k) {
        int nt = ntBase + k;
        v8f acc = {};
        acc = wmma_f16(a, (k == 0) ? bA : bB, acc);
        for (int r = 0; r < 8; ++r) {
          int i = mt * 16 + hi * 8 + r;
          int j = nt * 16 + n;
          S[i * 64 + j] = acc[r] * SCALE + biasG[(size_t)h * 4096 + i * 64 + j];
        }
      }
    }
    __syncthreads();

    // row softmax (f32 in LDS) -> probabilities f16
    if (tid < 64) {
      float mx = -3.4e38f;
      for (int j = 0; j < 64; ++j) mx = fmaxf(mx, S[tid * 64 + j]);
      float sum = 0.f;
      for (int j = 0; j < 64; ++j) sum += __expf(S[tid * 64 + j] - mx);
      float inv = 1.f / sum;
      for (int j = 0; j < 64; ++j)
        P[tid * 64 + j] = (_Float16)(__expf(S[tid * 64 + j] - mx) * inv);
    }
    __syncthreads();

    // O = P @ V : 4x2 tiles, K=64 -> 2 WMMAs; one tile per wave
    {
      int mt = wid >> 1, nt = wid & 1;
      v8f acc = {};
      acc = wmma_f16(ldA_frag(P, 64, mt * 16, 0),  ldB_frag(Vt, 64, nt * 16, 0),  acc);
      acc = wmma_f16(ldA_frag(P, 64, mt * 16, 32), ldB_frag(Vt, 64, nt * 16, 32), acc);
      for (int r = 0; r < 8; ++r) {
        int tok = mt * 16 + hi * 8 + r;
        int c = h * 32 + nt * 16 + n;
        int ph = wy * 8 + (tok >> 3), pw = wx * 8 + (tok & 7);
        attnh[(((size_t)b * 96 + c) << 16) + (ph << 8) + pw] = (_Float16)acc[r];
      }
    }
  }
}

// ---------------------------------------------------------------------------
// K5: depthwise 5x5 (reflect pad 2) + attn add -> U in GEMM layout [pix][96].
// 16x16 spatial tile + 2-halo staged in LDS.  Reference bug preserved: BOTH
// branches add the MAIN attention output.
// ---------------------------------------------------------------------------
__global__ __launch_bounds__(256) void dwconv_merge(
    const _Float16* __restrict__ vh, const _Float16* __restrict__ avh,
    const _Float16* __restrict__ attnh, const float* __restrict__ wdw,
    const float* __restrict__ bdw, const float* __restrict__ wdwa,
    const float* __restrict__ bdwa, _Float16* __restrict__ u0,
    _Float16* __restrict__ u1) {
  __shared__ _Float16 tileS[20 * 20];
  int tid = threadIdx.x;
  int bid = blockIdx.x;
  const int blocksPerTensor = NB * 96 * 256;   // 98304
  int which = bid >= blocksPerTensor;
  int r = which ? bid - blocksPerTensor : bid;
  int bc = r >> 8;                             // b*96 + c
  int tile = r & 255;
  int y0 = (tile >> 4) * 16, x0 = (tile & 15) * 16;
  int c = bc % 96;

  const _Float16* plane = (which ? avh : vh) + ((size_t)bc << 16);
  for (int e = tid; e < 400; e += 256) {
    int ly = e / 20, lx = e - ly * 20;
    int gy = y0 + ly - 2; gy = gy < 0 ? -gy : (gy > 255 ? 510 - gy : gy);
    int gx = x0 + lx - 2; gx = gx < 0 ? -gx : (gx > 255 ? 510 - gx : gx);
    tileS[e] = plane[(gy << 8) + gx];
  }
  __syncthreads();

  const float* wk = (which ? wdwa : wdw) + c * 25;
  float acc = (which ? bdwa : bdw)[c];
  int oy = tid >> 4, ox = tid & 15;
#pragma unroll
  for (int ky = 0; ky < 5; ++ky)
#pragma unroll
    for (int kx = 0; kx < 5; ++kx)
      acc += (float)tileS[(oy + ky) * 20 + (ox + kx)] * wk[ky * 5 + kx];

  int hh = y0 + oy, ww = x0 + ox;
  acc += (float)attnh[((size_t)bc << 16) + (hh << 8) + ww];
  int b = bc / 96;
  size_t pix = ((size_t)b << 16) + (hh << 8) + ww;
  (which ? u1 : u0)[pix * 96 + c] = (_Float16)acc;
}

// ---------------------------------------------------------------------------
// K6: final conv1x1: [262144 x 96] x [96 x 96] WMMA GEMM per branch, f32 out.
// A-fragments hoisted per wave; B-fragments software-pipelined across nt.
// ---------------------------------------------------------------------------
__global__ __launch_bounds__(256) void out_gemm(
    const _Float16* __restrict__ u0, const _Float16* __restrict__ u1,
    const float* __restrict__ wp, const float* __restrict__ bp,
    const float* __restrict__ wpa, const float* __restrict__ bpa,
    float* __restrict__ out) {
  __shared__ __align__(16) _Float16 sa[128 * 96];
  __shared__ __align__(16) _Float16 sw[96 * 96];
  __shared__ float sb[96];
  int which = blockIdx.y;
  const _Float16* U = which ? u1 : u0;
  const float* wgt = which ? wpa : wp;
  const float* bia = which ? bpa : bp;
  float* outT = out + (size_t)which * NB * 96 * HW;
  int pix0 = blockIdx.x * 128;
  int tid = threadIdx.x;

  {
    const uint4* src = (const uint4*)(U + (size_t)pix0 * 96);
    uint4* dst = (uint4*)sa;
#if HAVE_ASYNC_LDS
    for (int e = tid; e < 1536; e += 256) async_copy_b128(dst + e, src + e);
#else
    for (int e = tid; e < 1536; e += 256) dst[e] = src[e];
#endif
  }
  for (int e = tid; e < 96 * 96; e += 256) sw[e] = (_Float16)wgt[e];
  if (tid < 96) sb[tid] = bia[tid];
#if HAVE_ASYNC_LDS
  wait_async();
#endif
  __syncthreads();

  int wid = tid >> 5, lane = tid & 31, n = lane & 15, hi = lane >> 4;
  int b = pix0 >> 16;
  int mt = wid;                                 // 8 waves x 8 m-tiles
  v16h a0 = ldA_frag(sa, 96, mt * 16, 0);
  v16h a1 = ldA_frag(sa, 96, mt * 16, 32);
  v16h a2 = ldA_frag(sa, 96, mt * 16, 64);
  v16h b0 = ldB_frag(sw, 96, 0, 0);
  v16h b1 = ldB_frag(sw, 96, 0, 32);
  v16h b2 = ldB_frag(sw, 96, 0, 64);
#pragma unroll
  for (int nt = 0; nt < 6; ++nt) {
    v16h c0 = b0, c1 = b1, c2 = b2;
    if (nt < 5) {
      b0 = ldB_frag(sw, 96, (nt + 1) * 16, 0);
      b1 = ldB_frag(sw, 96, (nt + 1) * 16, 32);
      b2 = ldB_frag(sw, 96, (nt + 1) * 16, 64);
    }
    v8f acc = {};
    acc = wmma_f16(a0, c0, acc);
    acc = wmma_f16(a1, c1, acc);
    acc = wmma_f16(a2, c2, acc);
    int oc = nt * 16 + n;
    float bias = sb[oc];
    for (int r = 0; r < 8; ++r) {
      int p = pix0 + mt * 16 + hi * 8 + r;
      int rem = p & (HW - 1);
      outT[(((size_t)b * 96 + oc) << 16) + rem] = acc[r] + bias;
    }
  }
}

// ---------------------------------------------------------------------------
// Host launcher
// ---------------------------------------------------------------------------
extern "C" void kernel_launch(void* const* d_in, const int* in_sizes, int n_in,
                              void* d_out, int out_size, void* d_ws, size_t ws_size,
                              hipStream_t stream) {
  const float* vision     = (const float*)d_in[0];
  const float* ass_vision = (const float*)d_in[1];
  const float* wv   = (const float*)d_in[2];
  const float* bv   = (const float*)d_in[3];
  const float* wav  = (const float*)d_in[4];
  const float* bav  = (const float*)d_in[5];
  const float* wqk  = (const float*)d_in[6];
  const float* bqk  = (const float*)d_in[7];
  // d_in[8] = waqk, d_in[9] = baqk : unused (reference discards ass attention)
  const float* wdw  = (const float*)d_in[10];
  const float* bdw  = (const float*)d_in[11];
  const float* wdwa = (const float*)d_in[12];
  const float* bdwa = (const float*)d_in[13];
  const float* wp   = (const float*)d_in[14];
  const float* bp   = (const float*)d_in[15];
  const float* wpa  = (const float*)d_in[16];
  const float* bpa  = (const float*)d_in[17];
  const float* mw1  = (const float*)d_in[18];
  const float* mb1  = (const float*)d_in[19];
  const float* mw2  = (const float*)d_in[20];
  const float* mb2  = (const float*)d_in[21];

  const size_t PLANE_H = (size_t)NB * 96 * HW * 2;        // 50,331,648 B
  char* w = (char*)d_ws;
  _Float16* vh    = (_Float16*)(w);
  _Float16* avh   = (_Float16*)(w + PLANE_H);
  _Float16* qkvh  = (_Float16*)(w + 2 * PLANE_H);          // [4096,64,288] f16
  _Float16* attnh = (_Float16*)(w + 2 * PLANE_H + (size_t)NWIN * 64 * C3 * 2);
  float*    biasb = (float*)   (w + 2 * PLANE_H + (size_t)NWIN * 64 * C3 * 2 + PLANE_H);
  // QKV buffer is dead after win_attn -> reuse for U tensors
  _Float16* u0 = qkvh;
  _Float16* u1 = (_Float16*)((char*)qkvh + PLANE_H);

  bias_mlp<<<16, 256, 0, stream>>>(mw1, mb1, mw2, mb2, biasb);
  conv_qkv_main<<<NWIN, 256, 0, stream>>>(vision, wqk, bqk, wv, bv, qkvh, vh);
  conv_v_ass<<<NWIN, 256, 0, stream>>>(ass_vision, wav, bav, avh);
  win_attn<<<NWIN, 256, 0, stream>>>(qkvh, biasb, attnh);
  dwconv_merge<<<2 * NB * 96 * 256, 256, 0, stream>>>(
      vh, avh, attnh, wdw, bdw, wdwa, bdwa, u0, u1);
  out_gemm<<<dim3(2048, 2), 256, 0, stream>>>(u0, u1, wp, bp, wpa, bpa,
                                              (float*)d_out);
}
